// CosineTripletLoss_69028714381658
// MI455X (gfx1250) — matrix operand order repaired and checked by reading, not compiled, not run on hardware
//
#include <hip/hip_runtime.h>

// ---------------------------------------------------------------------------
// CosineTripletLoss for MI455X (gfx1250, wave32, WMMA)
//   sim GEMM (8192x8192x1024) fused with masked row-argmax, run on
//   v_wmma_f32_16x16x32_bf16 with a bf16 hi/lo compensated split (bf16x3)
//   for ~fp32-accurate mining. pos/neg sims and the loss are exact fp32.
//   Each wave owns a 32x32 register tile (4 accumulators) to balance the
//   LDS path (ds_load_b128) against the XDL WMMA pipes.
// ---------------------------------------------------------------------------

typedef __attribute__((ext_vector_type(16))) __bf16 v16bf;
typedef __attribute__((ext_vector_type(8)))  float  v8f;

#define NROWS 8192
#define DDIM  1024
#define BM 64
#define BN 128
#define BK 64
#define MARGIN 0.05f

union Frag {
  uint4 q[2];
  v16bf v;
};

__device__ inline v8f wmma_bf16(v16bf a, v16bf b, v8f c) {
  // 8 args: (neg_a, A, neg_b, B, c_mod, C, reuse_a, reuse_b)
  return __builtin_amdgcn_wmma_f32_16x16x32_bf16(
      false, a, false, b, (short)0, c, false, false);
}

// round-to-nearest-even float -> bf16 bits, plus residual split
__device__ inline unsigned short f2bf_rne(float f) {
  unsigned u = __float_as_uint(f);
  u += 0x7fffu + ((u >> 16) & 1u);
  return (unsigned short)(u >> 16);
}
__device__ inline void split_bf16(float f, unsigned short& h, unsigned short& l) {
  unsigned u = __float_as_uint(f);
  unsigned uh = u + (0x7fffu + ((u >> 16) & 1u));
  unsigned short hb = (unsigned short)(uh >> 16);
  float fh = __uint_as_float(((unsigned)hb) << 16);
  h = hb;
  l = f2bf_rne(f - fh);
}

// --------------------------------------------------------------------------
// Kernel 1: per-row fp32 pos_sim + bf16 hi/lo conversion of x and y
// grid = NROWS blocks, 256 threads (4 elems/thread)
// --------------------------------------------------------------------------
__global__ __launch_bounds__(256) void prep_kernel(
    const float* __restrict__ x, const float* __restrict__ y,
    unsigned short* __restrict__ xh, unsigned short* __restrict__ xl,
    unsigned short* __restrict__ yh, unsigned short* __restrict__ yl,
    float* __restrict__ pos) {
  const int i = blockIdx.x;
  const int t = threadIdx.x;
  const float4 xv = ((const float4*)(x + (size_t)i * DDIM))[t];
  const float4 yv = ((const float4*)(y + (size_t)i * DDIM))[t];

  float d = xv.x * yv.x + xv.y * yv.y + xv.z * yv.z + xv.w * yv.w;

  ushort4 xhv, xlv, yhv, ylv;
  split_bf16(xv.x, xhv.x, xlv.x);
  split_bf16(xv.y, xhv.y, xlv.y);
  split_bf16(xv.z, xhv.z, xlv.z);
  split_bf16(xv.w, xhv.w, xlv.w);
  split_bf16(yv.x, yhv.x, ylv.x);
  split_bf16(yv.y, yhv.y, ylv.y);
  split_bf16(yv.z, yhv.z, ylv.z);
  split_bf16(yv.w, yhv.w, ylv.w);
  ((ushort4*)(xh + (size_t)i * DDIM))[t] = xhv;
  ((ushort4*)(xl + (size_t)i * DDIM))[t] = xlv;
  ((ushort4*)(yh + (size_t)i * DDIM))[t] = yhv;
  ((ushort4*)(yl + (size_t)i * DDIM))[t] = ylv;

  // block reduction (8 waves of 32)
  for (int m = 16; m >= 1; m >>= 1) d += __shfl_xor(d, m, 32);
  __shared__ float sw[8];
  if ((t & 31) == 0) sw[t >> 5] = d;
  __syncthreads();
  if (t == 0) {
    float s = 0.f;
    for (int w = 0; w < 8; ++w) s += sw[w];
    pos[i] = s;
  }
}

// --------------------------------------------------------------------------
// Kernel 2: fused masked-argmax GEMM (hard-negative mining)
// grid = NROWS/BM = 128 blocks, 256 threads = 8 waves.
// Waves laid out 2(m) x 4(n); each wave owns a 32x32 output tile
// (2 m-tiles x 2 n-tiles, 4 fp32 accumulators).
// bf16x3: acc += Ah*Bh + Ah*Bl + Al*Bh  (single fp32 accumulator chain)
// --------------------------------------------------------------------------
__device__ inline void loadA(const uint4* S, int row_local, int s, int g, Frag& f) {
  // A 16x32 bf16 layout: lane half g holds K = 8g+{0..7} and 8g+16+{0..7}
  int q = row_local * (BK / 8) + s * 4 + g;
  f.q[0] = S[q];
  f.q[1] = S[q + 2];
}
__device__ inline void loadB(const uint4* S, int col_local, int s, int g, Frag& f) {
  // B 32x16 bf16 layout: lane half g holds K = 8g+{0..7} and 8g+16+{0..7}
  int q = col_local * (BK / 8) + s * 4 + g;
  f.q[0] = S[q];
  f.q[1] = S[q + 2];
}

__global__ __launch_bounds__(256) void mine_kernel(
    const unsigned short* __restrict__ xh, const unsigned short* __restrict__ xl,
    const unsigned short* __restrict__ yh, const unsigned short* __restrict__ yl,
    const float* __restrict__ pos, int* __restrict__ neg_idx) {
  __shared__ unsigned short sXh[BM * BK], sXl[BM * BK];   // 8 KB each
  __shared__ unsigned short sYh[BN * BK], sYl[BN * BK];   // 16 KB each
  __shared__ float sPos[BM];
  __shared__ float sRv[BM * 4];
  __shared__ int   sRi[BM * 4];

  const int tid = threadIdx.x;
  const int lane = tid & 31;
  const int wave = tid >> 5;
  const int half = lane >> 4;   // K-half selector per WMMA layout
  const int l16  = lane & 15;
  const int mh = wave >> 2;     // m-half 0..1  -> rows mh*32 .. +31
  const int ng = wave & 3;      // n-group 0..3 -> cols ng*32 .. +31
  const int row_base = blockIdx.x * BM;

  if (tid < BM) sPos[tid] = pos[row_base + tid];
  __syncthreads();

  // pos_sim for this wave's 16 row-slots (2 m-tiles x 8 C-rows)
  float posr[2][8];
#pragma unroll
  for (int mi = 0; mi < 2; ++mi)
#pragma unroll
    for (int r = 0; r < 8; ++r)
      posr[mi][r] = sPos[mh * 32 + mi * 16 + r + 8 * half];

  float bv[2][8];
  int   bi[2][8];
#pragma unroll
  for (int mi = 0; mi < 2; ++mi)
#pragma unroll
    for (int r = 0; r < 8; ++r) { bv[mi][r] = -2.0f; bi[mi][r] = 0; }

  const unsigned short* xh_blk = xh + (size_t)row_base * DDIM;
  const unsigned short* xl_blk = xl + (size_t)row_base * DDIM;

  for (int cb = 0; cb < NROWS / BN; ++cb) {
    const int col_base = cb * BN;
    const unsigned short* yh_blk = yh + (size_t)col_base * DDIM;
    const unsigned short* yl_blk = yl + (size_t)col_base * DDIM;

    v8f acc[2][2];
#pragma unroll
    for (int mi = 0; mi < 2; ++mi)
#pragma unroll
      for (int ni = 0; ni < 2; ++ni)
#pragma unroll
        for (int r = 0; r < 8; ++r) acc[mi][ni][r] = 0.f;

    for (int k0 = 0; k0 < DDIM; k0 += BK) {
      __syncthreads();
      // ---- stage x tile: 512 uint4 per split (2 chunks/thread) ----
#pragma unroll
      for (int it = 0; it < 2; ++it) {
        int c = tid + it * 256;
        int rw = c >> 3, ko = (c & 7) << 3;
        size_t g = (size_t)rw * DDIM + k0 + ko;
        ((uint4*)sXh)[c] = *(const uint4*)(xh_blk + g);
        ((uint4*)sXl)[c] = *(const uint4*)(xl_blk + g);
        if (k0 + BK < DDIM) {
          __builtin_prefetch(xh_blk + g + BK, 0, 0);
          __builtin_prefetch(xl_blk + g + BK, 0, 0);
        }
      }
      // ---- stage y tile: 1024 uint4 per split (4 chunks/thread) ----
#pragma unroll
      for (int it = 0; it < 4; ++it) {
        int c = tid + it * 256;
        int rw = c >> 3, ko = (c & 7) << 3;
        size_t g = (size_t)rw * DDIM + k0 + ko;
        ((uint4*)sYh)[c] = *(const uint4*)(yh_blk + g);
        ((uint4*)sYl)[c] = *(const uint4*)(yl_blk + g);
        if (k0 + BK < DDIM) {
          __builtin_prefetch(yh_blk + g + BK, 0, 0);
          __builtin_prefetch(yl_blk + g + BK, 0, 0);
        }
      }
      __syncthreads();

#pragma unroll
      for (int s = 0; s < 2; ++s) {   // two K=32 steps per K=64 stage
        Frag a[2][2], b[2][2];        // [tile][hi/lo]
#pragma unroll
        for (int mi = 0; mi < 2; ++mi) {
          int rloc = (mh * 2 + mi) * 16 + l16;
          loadA((const uint4*)sXh, rloc, s, half, a[mi][0]);
          loadA((const uint4*)sXl, rloc, s, half, a[mi][1]);
        }
#pragma unroll
        for (int ni = 0; ni < 2; ++ni) {
          int cloc = (ng * 2 + ni) * 16 + l16;
          loadB((const uint4*)sYh, cloc, s, half, b[ni][0]);
          loadB((const uint4*)sYl, cloc, s, half, b[ni][1]);
        }
#pragma unroll
        for (int mi = 0; mi < 2; ++mi)
#pragma unroll
          for (int ni = 0; ni < 2; ++ni) {
            acc[mi][ni] = wmma_bf16(a[mi][0].v, b[ni][0].v, acc[mi][ni]);
            acc[mi][ni] = wmma_bf16(a[mi][0].v, b[ni][1].v, acc[mi][ni]);
            acc[mi][ni] = wmma_bf16(a[mi][1].v, b[ni][0].v, acc[mi][ni]);
          }
      }
    }

    // mask (diag or sim > pos_sim -> -1.0) and fold into running argmax
#pragma unroll
    for (int mi = 0; mi < 2; ++mi) {
      int rl0 = mh * 32 + mi * 16 + 8 * half;
#pragma unroll
      for (int r = 0; r < 8; ++r) {
        const int irow = row_base + rl0 + r;
        const float p = posr[mi][r];
#pragma unroll
        for (int ni = 0; ni < 2; ++ni) {
          int j = col_base + (ng * 2 + ni) * 16 + l16;
          float v = acc[mi][ni][r];
          if (j == irow || v > p) v = -1.0f;
          if (v > bv[mi][r] || (v == bv[mi][r] && j < bi[mi][r])) {
            bv[mi][r] = v;
            bi[mi][r] = j;
          }
        }
      }
    }
  }

  // reduce across the 16 lanes (same row, different N) with first-index ties
#pragma unroll
  for (int mi = 0; mi < 2; ++mi)
#pragma unroll
    for (int r = 0; r < 8; ++r) {
      float v = bv[mi][r];
      int   j = bi[mi][r];
      for (int m = 1; m <= 8; m <<= 1) {
        float ov = __shfl_xor(v, m, 32);
        int   oj = __shfl_xor(j, m, 32);
        if (ov > v || (ov == v && oj < j)) { v = ov; j = oj; }
      }
      if (l16 == 0) {
        int rl = mh * 32 + mi * 16 + r + 8 * half;
        sRv[rl * 4 + ng] = v;
        sRi[rl * 4 + ng] = j;
      }
    }
  __syncthreads();
  if (tid < BM) {
    float v = sRv[tid * 4];
    int   j = sRi[tid * 4];
#pragma unroll
    for (int g = 1; g < 4; ++g) {
      float ov = sRv[tid * 4 + g];
      int   oj = sRi[tid * 4 + g];
      if (ov > v || (ov == v && oj < j)) { v = ov; j = oj; }
    }
    neg_idx[row_base + tid] = j;
  }
}

// --------------------------------------------------------------------------
// Kernel 3: exact fp32 neg_sim + per-row hinge loss
// --------------------------------------------------------------------------
__global__ __launch_bounds__(256) void loss_kernel(
    const float* __restrict__ x, const float* __restrict__ y,
    const float* __restrict__ pos, const int* __restrict__ neg_idx,
    float* __restrict__ loss) {
  const int i = blockIdx.x;
  const int t = threadIdx.x;
  const int nj = neg_idx[i];
  const float4 xv = ((const float4*)(x + (size_t)i * DDIM))[t];
  const float4 yv = ((const float4*)(y + (size_t)nj * DDIM))[t];
  float d = xv.x * yv.x + xv.y * yv.y + xv.z * yv.z + xv.w * yv.w;
  for (int m = 16; m >= 1; m >>= 1) d += __shfl_xor(d, m, 32);
  __shared__ float sw[8];
  if ((t & 31) == 0) sw[t >> 5] = d;
  __syncthreads();
  if (t == 0) {
    float s = 0.f;
    for (int w = 0; w < 8; ++w) s += sw[w];
    float li = MARGIN - pos[i] + s;
    loss[i] = li > 0.f ? li : 0.f;
  }
}

// --------------------------------------------------------------------------
// Kernel 4: mean over rows -> scalar (overwrite d_out; replay-safe)
// --------------------------------------------------------------------------
__global__ __launch_bounds__(256) void reduce_kernel(
    const float* __restrict__ loss, float* __restrict__ out) {
  float s = 0.f;
  for (int k = threadIdx.x; k < NROWS; k += 256) s += loss[k];
  for (int m = 16; m >= 1; m >>= 1) s += __shfl_xor(s, m, 32);
  __shared__ float sw[8];
  if ((threadIdx.x & 31) == 0) sw[threadIdx.x >> 5] = s;
  __syncthreads();
  if (threadIdx.x == 0) {
    float t = 0.f;
    for (int w = 0; w < 8; ++w) t += sw[w];
    out[0] = t * (1.0f / (float)NROWS);
  }
}

// --------------------------------------------------------------------------
extern "C" void kernel_launch(void* const* d_in, const int* in_sizes, int n_in,
                              void* d_out, int out_size, void* d_ws, size_t ws_size,
                              hipStream_t stream) {
  (void)in_sizes; (void)n_in; (void)out_size; (void)ws_size;
  const float* x = (const float*)d_in[0];
  const float* y = (const float*)d_in[1];

  char* ws = (char*)d_ws;
  const size_t half_bytes = (size_t)NROWS * DDIM * sizeof(unsigned short); // 16 MB
  unsigned short* xh = (unsigned short*)(ws + 0 * half_bytes);
  unsigned short* xl = (unsigned short*)(ws + 1 * half_bytes);
  unsigned short* yh = (unsigned short*)(ws + 2 * half_bytes);
  unsigned short* yl = (unsigned short*)(ws + 3 * half_bytes);
  float* pos  = (float*)(ws + 4 * half_bytes);
  int*   neg  = (int*)(ws + 4 * half_bytes + NROWS * sizeof(float));
  float* loss = (float*)(ws + 4 * half_bytes + NROWS * (sizeof(float) + sizeof(int)));
  float* out  = (float*)d_out;

  prep_kernel<<<NROWS, 256, 0, stream>>>(x, y, xh, xl, yh, yl, pos);
  mine_kernel<<<NROWS / BM, 256, 0, stream>>>(xh, xl, yh, yl, pos, neg);
  loss_kernel<<<NROWS, 256, 0, stream>>>(x, y, pos, neg, loss);
  reduce_kernel<<<1, 256, 0, stream>>>(loss, out);
}
